// T3STAEformer_70841190580660
// MI455X (gfx1250) — compile-verified, matrix operand order checked by reading, not run on hardware
//
#include <hip/hip_runtime.h>
#include <hip/hip_bf16.h>

// ---------------------------------------------------------------------------
// T3 STAEformer for MI455X (gfx1250, wave32).
// GEMMs (Q/K/V/O projections + FF) run on v_wmma_f32_16x16x32_f16 with a
// software-pipelined, double-buffered LDS scheme: global b128 loads for the
// next K-slab are in flight while the current slab's fragments are read from
// LDS (ds_load_b128) and fed to 4 WMMAs per wave (A-fragment reuse across a
// 16x64 wave tile). One barrier per K-step.
// Attention softmax, LayerNorm, gathers and output heads stay in f32 VALU.
// Workspace requirement: ~321 MB (6 x 42MB token buffers + 1 x 70MB FF buf).
// ---------------------------------------------------------------------------

#define B_     16
#define L_     12
#define N_     358
#define D_     152
#define HEADS_ 4
#define HD_    38
#define FF_    256
#define PRED_  12
#define TOK_   (B_ * L_ * N_)      // 68736
#define EVTD_  1024
#define HEVT_  256
#define KMAX_  8
#define FUSED_ (HEVT_ + D_)        // 408

#define GT_M 128   // GEMM macro-tile M (TOK_ = 537 * 128, exact)
#define GT_N 64    // GEMM macro-tile N (wave tile = 16 x 64 -> 4 WMMAs/step)
#define GT_K 32    // GEMM K step (one WMMA depth)

typedef __attribute__((ext_vector_type(16))) _Float16 v16h;
typedef __attribute__((ext_vector_type(8)))  float    v8f;

// ---------------------------------------------------------------------------
// Embedding: x[b,l,n,:] = concat(x_in@W_in+b_in, tod_emb[tod], dow_emb[dow],
//                                adaptive_emb[l,n])
// ---------------------------------------------------------------------------
__global__ void embed_kernel(const float* __restrict__ var_x,
                             const float* __restrict__ marker,
                             const float* __restrict__ W_in,
                             const float* __restrict__ b_in,
                             const float* __restrict__ tod_emb,
                             const float* __restrict__ dow_emb,
                             const float* __restrict__ adaptive,
                             float* __restrict__ x) {
  long gid = (long)blockIdx.x * blockDim.x + threadIdx.x;
  if (gid >= (long)TOK_ * D_) return;
  int d = (int)(gid % D_);
  long tok = gid / D_;
  int n = (int)(tok % N_);
  int l = (int)((tok / N_) % L_);
  const float* mk = marker + tok * 4;
  float out;
  if (d < 24) {
    float x0 = var_x[tok], x1 = mk[0], x2 = mk[1];
    out = x0 * W_in[0 * 24 + d] + x1 * W_in[1 * 24 + d] + x2 * W_in[2 * 24 + d]
        + b_in[d];
  } else if (d < 48) {
    int tod = (int)(mk[0] * 144.0f);
    tod = tod < 0 ? 0 : (tod > 143 ? 143 : tod);
    out = tod_emb[tod * 24 + (d - 24)];
  } else if (d < 72) {
    int dow = (int)(mk[1] * 7.0f);
    dow = dow < 0 ? 0 : (dow > 6 ? 6 : dow);
    out = dow_emb[dow * 24 + (d - 48)];
  } else {
    out = adaptive[((long)l * N_ + n) * 80 + (d - 72)];
  }
  x[tok * D_ + d] = out;
}

// ---------------------------------------------------------------------------
// WMMA GEMM, double-buffered LDS pipeline: C[M,N] = act(A[M,K]@W[K,N]+bias).
// Block (256 thr / 8 waves) computes a 128x64 C macro-tile. Wave w owns rows
// [w*16, w*16+16) x all 64 columns -> 4 accumulators, A-fragment reused 4x.
// A tile staged row-major f16 [128][32]; B tile staged K-transposed [64][32]
// so fragment reads are contiguous 16B runs (ds_load_b128).
// Fragment layouts per ISA 7.12.2 (16-bit, wave32):
//   lane = 16*half + l16; elements 0..7 -> K = half*8 + {0..7},
//   elements 8..15 -> K = 16 + half*8 + {0..7}.
//   C: VGPR r -> M = half*8 + r, N = l16.
// Pipeline: ds_store(buf) -> issue global loads slab+1 -> barrier ->
//           ds_load frags + 4 WMMAs from buf -> flip buf.
// ---------------------------------------------------------------------------
__global__ void __launch_bounds__(256)
wmma_gemm_kernel(const float* __restrict__ A, const float* __restrict__ W,
                 const float* __restrict__ bias, float* __restrict__ C,
                 int M, int N, int K, int relu) {
  __shared__ _Float16 As[2][GT_M][GT_K];  // 16 KB
  __shared__ _Float16 Bs[2][GT_N][GT_K];  // 8 KB (K-transposed)

  const int tid  = threadIdx.x;
  const int wave = tid >> 5;
  const int lane = tid & 31;
  const int half = lane >> 4;
  const int l16  = lane & 15;
  const int kb   = half * 8;

  const int tilesN = (N + GT_N - 1) / GT_N;
  const int row0 = (blockIdx.x / tilesN) * GT_M;
  const int col0 = (blockIdx.x % tilesN) * GT_N;
  const bool interiorMN = (row0 + GT_M <= M) && (col0 + GT_N <= N);

  // --- slab loader: A 128x32 (4 float4/thr), B 32x64 (2 float4/thr) ---
  auto loadSlab = [&](int k0c, float4 (&va)[4], float4 (&vb)[2]) {
    const bool inter = interiorMN && (k0c + GT_K <= K);
    if (inter) {
#pragma unroll
      for (int i = 0; i < 4; ++i) {
        int s  = tid + i * 256;        // A float4 slot 0..1023
        int r  = s >> 3;               // 0..127
        int kq = (s & 7) << 2;         // 0..28
        va[i] = *reinterpret_cast<const float4*>(
            A + (long)(row0 + r) * K + (k0c + kq));
      }
#pragma unroll
      for (int i = 0; i < 2; ++i) {
        int s  = tid + i * 256;        // B float4 slot 0..511
        int kB = s >> 4;               // 0..31
        int nq = (s & 15) << 2;        // 0..60
        vb[i] = *reinterpret_cast<const float4*>(
            W + (long)(k0c + kB) * N + (col0 + nq));
      }
    } else {
#pragma unroll
      for (int i = 0; i < 4; ++i) {
        int s  = tid + i * 256;
        int r  = s >> 3;
        int kq = (s & 7) << 2;
        int gr = row0 + r, gk = k0c + kq;
        va[i] = make_float4(0.f, 0.f, 0.f, 0.f);
        if (gr < M && gk < K)          // K multiple of 4 -> whole-vec guard
          va[i] = *reinterpret_cast<const float4*>(A + (long)gr * K + gk);
      }
#pragma unroll
      for (int i = 0; i < 2; ++i) {
        int s  = tid + i * 256;
        int kB = s >> 4;
        int nq = (s & 15) << 2;
        int gk = k0c + kB, gn = col0 + nq;
        vb[i] = make_float4(0.f, 0.f, 0.f, 0.f);
        if (gk < K && gn < N)          // N multiple of 4 -> whole-vec guard
          vb[i] = *reinterpret_cast<const float4*>(W + (long)gk * N + gn);
      }
    }
  };
  // --- slab -> LDS (f32 -> f16 conversion at store) ---
  auto storeSlab = [&](int bufc, const float4 (&va)[4], const float4 (&vb)[2]) {
#pragma unroll
    for (int i = 0; i < 4; ++i) {
      int s  = tid + i * 256;
      int r  = s >> 3;
      int kq = (s & 7) << 2;
      As[bufc][r][kq + 0] = (_Float16)va[i].x;
      As[bufc][r][kq + 1] = (_Float16)va[i].y;
      As[bufc][r][kq + 2] = (_Float16)va[i].z;
      As[bufc][r][kq + 3] = (_Float16)va[i].w;
    }
#pragma unroll
    for (int i = 0; i < 2; ++i) {
      int s  = tid + i * 256;
      int kB = s >> 4;
      int nq = (s & 15) << 2;
      Bs[bufc][nq + 0][kB] = (_Float16)vb[i].x;
      Bs[bufc][nq + 1][kB] = (_Float16)vb[i].y;
      Bs[bufc][nq + 2][kB] = (_Float16)vb[i].z;
      Bs[bufc][nq + 3][kB] = (_Float16)vb[i].w;
    }
  };

  v8f acc[4] = {};
  float4 va[4];
  float4 vb[2];
  loadSlab(0, va, vb);  // prologue
  int buf = 0;

  for (int k0 = 0; k0 < K; k0 += GT_K) {
    storeSlab(buf, va, vb);
    if (k0 + GT_K < K) loadSlab(k0 + GT_K, va, vb);  // in flight during WMMA
    // warm L2 for the slab after next (global_prefetch_b8)
    if (k0 + 2 * GT_K < K && tid < GT_M)
      __builtin_prefetch(A + (long)(row0 + tid) * K + (k0 + 2 * GT_K), 0, 1);
    __syncthreads();

    // ---- fragments from LDS (contiguous 16B runs -> ds_load_b128) ----
    const int ar = wave * 16 + l16;
    v16h a;
#pragma unroll
    for (int e = 0; e < 8; ++e) {
      a[e]     = As[buf][ar][kb + e];
      a[e + 8] = As[buf][ar][16 + kb + e];
    }
#pragma unroll
    for (int nt = 0; nt < 4; ++nt) {
      v16h bv;
#pragma unroll
      for (int e = 0; e < 8; ++e) {
        bv[e]     = Bs[buf][nt * 16 + l16][kb + e];
        bv[e + 8] = Bs[buf][nt * 16 + l16][16 + kb + e];
      }
      acc[nt] = __builtin_amdgcn_wmma_f32_16x16x32_f16(
          false, a, false, bv, (short)0, acc[nt], false, false);
    }
    buf ^= 1;
  }

  // ---- store C ----
#pragma unroll
  for (int nt = 0; nt < 4; ++nt) {
    int n = col0 + nt * 16 + l16;
    if (n >= N) continue;
    float bv = bias ? bias[n] : 0.0f;
#pragma unroll
    for (int r = 0; r < 8; ++r) {
      int m = row0 + wave * 16 + kb + r;
      if (m < M) {
        float v = acc[nt][r] + bv;
        if (relu) v = fmaxf(v, 0.0f);
        C[(long)m * N + n] = v;
      }
    }
  }
}

// ---------------------------------------------------------------------------
// Multi-head attention (online softmax), layout-parameterized.
// x layout is always (B, L, N, D). temporal: seq over L (stride N_ tokens),
// spatial: seq over N (stride 1 token). grid = (groups, HEADS_).
// ---------------------------------------------------------------------------
__global__ void attn_kernel(const float* __restrict__ Q,
                            const float* __restrict__ Km,
                            const float* __restrict__ Vm,
                            float* __restrict__ O, int temporal) {
  int g = blockIdx.x, h = blockIdx.y;
  long base;
  long stride;
  int S;
  if (temporal) {
    int b = g / N_, n = g % N_;
    S = L_;
    base = (long)b * L_ * N_ + n;
    stride = N_;
  } else {
    int b = g / L_, l = g % L_;
    S = N_;
    base = ((long)b * L_ + l) * N_;
    stride = 1;
  }
  const float scale = 0.1622214211f;  // 38^-0.5

  for (int qi = threadIdx.x; qi < S; qi += blockDim.x) {
    long qoff = (base + (long)qi * stride) * D_ + h * HD_;
    float qv[HD_];
#pragma unroll
    for (int d = 0; d < HD_; ++d) qv[d] = Q[qoff + d];
    float mx = -1e30f, lsum = 0.0f;
    float accv[HD_];
#pragma unroll
    for (int d = 0; d < HD_; ++d) accv[d] = 0.0f;
    for (int j = 0; j < S; ++j) {
      long koff = (base + (long)j * stride) * D_ + h * HD_;
      float s = 0.0f;
#pragma unroll
      for (int d = 0; d < HD_; ++d) s += qv[d] * Km[koff + d];
      s *= scale;
      float nm = fmaxf(mx, s);
      float corr = __expf(mx - nm);
      float p = __expf(s - nm);
      lsum = lsum * corr + p;
#pragma unroll
      for (int d = 0; d < HD_; ++d) accv[d] = accv[d] * corr + p * Vm[koff + d];
      mx = nm;
    }
    float inv = 1.0f / lsum;
#pragma unroll
    for (int d = 0; d < HD_; ++d) O[qoff + d] = accv[d] * inv;
  }
}

// ---------------------------------------------------------------------------
// x = LayerNorm(x + t) * g + b   (in-place on x; one wave per token)
// ---------------------------------------------------------------------------
__global__ void ln_residual_kernel(float* __restrict__ x,
                                   const float* __restrict__ t,
                                   const float* __restrict__ gw,
                                   const float* __restrict__ bw) {
  int tok = blockIdx.x * (blockDim.x >> 5) + (threadIdx.x >> 5);
  if (tok >= TOK_) return;
  int lane = threadIdx.x & 31;
  float vals[5];
  float s = 0.0f;
#pragma unroll
  for (int i = 0; i < 5; ++i) {
    int d = lane + i * 32;
    float v = 0.0f;
    if (d < D_) v = x[(long)tok * D_ + d] + t[(long)tok * D_ + d];
    vals[i] = v;
    s += v;
  }
#pragma unroll
  for (int m = 16; m; m >>= 1) s += __shfl_xor(s, m, 32);
  float mean = s * (1.0f / D_);
  float vs = 0.0f;
#pragma unroll
  for (int i = 0; i < 5; ++i) {
    int d = lane + i * 32;
    if (d < D_) {
      float c = vals[i] - mean;
      vs += c * c;
    }
  }
#pragma unroll
  for (int m = 16; m; m >>= 1) vs += __shfl_xor(vs, m, 32);
  float rs = rsqrtf(vs * (1.0f / D_) + 1e-5f);
#pragma unroll
  for (int i = 0; i < 5; ++i) {
    int d = lane + i * 32;
    if (d < D_)
      x[(long)tok * D_ + d] = (vals[i] - mean) * rs * gw[d] + bw[d];
  }
}

// ---------------------------------------------------------------------------
// Event path: gather KMAX event embeddings by day-of-year, 1024->256 linear,
// masked max-pool; fold the node-independent event part of W_mm into
// evt_part[b, p].
// ---------------------------------------------------------------------------
__global__ void event_kernel(const float* __restrict__ marker,
                             const float* __restrict__ E,
                             const float* __restrict__ W_evt,
                             const float* __restrict__ b_evt,
                             const int* __restrict__ doy_idx,
                             const unsigned char* __restrict__ doy_mask,
                             const float* __restrict__ W_mm,
                             float* __restrict__ evt_part) {
  __shared__ float evt_s[HEVT_];
  int b = blockIdx.x;
  int j = threadIdx.x;  // 256 threads: one output channel each
  float mval = marker[(((long)b * L_ + (L_ - 1)) * N_ + 0) * 4 + 3];
  int doy = (int)(mval * 365.0f);
  doy = doy < 0 ? 0 : (doy > 365 ? 365 : doy);
  float best = -1e30f;
  for (int kk = 0; kk < KMAX_; ++kk) {
    if (!doy_mask[doy * KMAX_ + kk]) continue;
    int id = doy_idx[doy * KMAX_ + kk];
    const float* e = E + (long)id * EVTD_;
    float s = b_evt[j];
    for (int t2 = 0; t2 < EVTD_; ++t2) s += e[t2] * W_evt[(long)t2 * HEVT_ + j];
    best = fmaxf(best, s);
  }
  evt_s[j] = best;
  __syncthreads();
  if (j < PRED_) {
    float s = 0.0f;
    for (int l = 0; l < L_; ++l)
      for (int c = 0; c < HEVT_; ++c)
        s += evt_s[c] * W_mm[((long)(l * FUSED_ + c)) * PRED_ + j];
    evt_part[b * PRED_ + j] = s;
  }
}

// ---------------------------------------------------------------------------
// Output heads: out[b,p,n] = evt_part[b,p] + b_mm[p] + b_ts[p]
//   + sum_{l,d} x[b,l,n,d] * (W_mm[l*408+256+d, p] + W_ts[l*152+d, p])
// ---------------------------------------------------------------------------
__global__ void final_kernel(const float* __restrict__ x,
                             const float* __restrict__ evt_part,
                             const float* __restrict__ W_mm,
                             const float* __restrict__ b_mm,
                             const float* __restrict__ W_ts,
                             const float* __restrict__ b_ts,
                             float* __restrict__ out) {
  int gid = blockIdx.x * blockDim.x + threadIdx.x;
  if (gid >= B_ * N_ * PRED_) return;
  int p = gid % PRED_;
  int n = (gid / PRED_) % N_;
  int b = gid / (PRED_ * N_);
  float s = evt_part[b * PRED_ + p] + b_mm[p] + b_ts[p];
  for (int l = 0; l < L_; ++l) {
    const float* ts = x + (((long)b * L_ + l) * N_ + n) * D_;
    const float* wm = W_mm + ((long)(l * FUSED_ + HEVT_)) * PRED_ + p;
    const float* wt = W_ts + ((long)(l * D_)) * PRED_ + p;
    for (int d = 0; d < D_; ++d)
      s += ts[d] * (wm[(long)d * PRED_] + wt[(long)d * PRED_]);
  }
  out[((long)b * PRED_ + p) * N_ + n] = s;
}

// ---------------------------------------------------------------------------
// Host-side orchestration (graph-capture safe: all launches on `stream`).
// ---------------------------------------------------------------------------
extern "C" void kernel_launch(void* const* d_in, const int* in_sizes, int n_in,
                              void* d_out, int out_size, void* d_ws,
                              size_t ws_size, hipStream_t stream) {
  (void)in_sizes; (void)n_in; (void)out_size; (void)ws_size;
  const float* var_x    = (const float*)d_in[0];
  const float* marker   = (const float*)d_in[1];
  const float* W_in     = (const float*)d_in[2];
  const float* b_in     = (const float*)d_in[3];
  const float* tod_emb  = (const float*)d_in[4];
  const float* dow_emb  = (const float*)d_in[5];
  const float* adaptive = (const float*)d_in[6];
  const float* Wq = (const float*)d_in[7];
  const float* bq = (const float*)d_in[8];
  const float* Wk = (const float*)d_in[9];
  const float* bk = (const float*)d_in[10];
  const float* Wv = (const float*)d_in[11];
  const float* bv = (const float*)d_in[12];
  const float* Wo = (const float*)d_in[13];
  const float* bo = (const float*)d_in[14];
  const float* W1 = (const float*)d_in[15];
  const float* b1 = (const float*)d_in[16];
  const float* W2 = (const float*)d_in[17];
  const float* b2 = (const float*)d_in[18];
  const float* ln1_g = (const float*)d_in[19];
  const float* ln1_b = (const float*)d_in[20];
  const float* ln2_g = (const float*)d_in[21];
  const float* ln2_b = (const float*)d_in[22];
  const float* E      = (const float*)d_in[23];
  const float* W_evt  = (const float*)d_in[24];
  const float* b_evt  = (const float*)d_in[25];
  const float* W_mm   = (const float*)d_in[26];
  const float* b_mm   = (const float*)d_in[27];
  const float* W_ts   = (const float*)d_in[28];
  const float* b_ts   = (const float*)d_in[29];
  const int*           doy_idx  = (const int*)d_in[30];
  const unsigned char* doy_mask = (const unsigned char*)d_in[31];

  float* ws = (float*)d_ws;
  const size_t XS = (size_t)TOK_ * D_;     // 10,447,872 floats
  float* x  = ws;                          // activations (B,L,N,D)
  float* q  = ws + 1 * XS;
  float* k  = ws + 2 * XS;
  float* v  = ws + 3 * XS;
  float* ao = ws + 4 * XS;                 // attention output
  float* t1 = ws + 5 * XS;                 // projection / FF2 temp (D wide)
  float* h1 = ws + 6 * XS;                 // FF1 temp (FF_ wide)
  float* evt_part = ws + 6 * XS + (size_t)TOK_ * FF_;  // (B, PRED)

  auto gemm = [&](const float* A, const float* W, const float* bias, float* C,
                  int M, int Nn, int Kk, int relu) {
    int tiles = ((M + GT_M - 1) / GT_M) * ((Nn + GT_N - 1) / GT_N);
    wmma_gemm_kernel<<<tiles, 256, 0, stream>>>(A, W, bias, C, M, Nn, Kk,
                                                relu);
  };

  // 1) embedding
  {
    long total = (long)TOK_ * D_;
    embed_kernel<<<(int)((total + 255) / 256), 256, 0, stream>>>(
        var_x, marker, W_in, b_in, tod_emb, dow_emb, adaptive, x);
  }

  // 2) six transformer layers (0-2 temporal over L, 3-5 spatial over N)
  for (int i = 0; i < 6; ++i) {
    int temporal = (i < 3) ? 1 : 0;
    const float* wq = Wq + (size_t)i * D_ * D_;
    const float* wk = Wk + (size_t)i * D_ * D_;
    const float* wv = Wv + (size_t)i * D_ * D_;
    const float* wo = Wo + (size_t)i * D_ * D_;

    gemm(x, wq, bq + i * D_, q, TOK_, D_, D_, 0);
    gemm(x, wk, bk + i * D_, k, TOK_, D_, D_, 0);
    gemm(x, wv, bv + i * D_, v, TOK_, D_, D_, 0);

    if (temporal) {
      attn_kernel<<<dim3(B_ * N_, HEADS_), 32, 0, stream>>>(q, k, v, ao, 1);
    } else {
      attn_kernel<<<dim3(B_ * L_, HEADS_), 128, 0, stream>>>(q, k, v, ao, 0);
    }

    gemm(ao, wo, bo + i * D_, t1, TOK_, D_, D_, 0);
    ln_residual_kernel<<<(TOK_ + 7) / 8, 256, 0, stream>>>(
        x, t1, ln1_g + i * D_, ln1_b + i * D_);

    gemm(x, W1 + (size_t)i * D_ * FF_, b1 + i * FF_, h1, TOK_, FF_, D_, 1);
    gemm(h1, W2 + (size_t)i * FF_ * D_, b2 + i * D_, t1, TOK_, D_, FF_, 0);
    ln_residual_kernel<<<(TOK_ + 7) / 8, 256, 0, stream>>>(
        x, t1, ln2_g + i * D_, ln2_b + i * D_);
  }

  // 3) event path (+ fold event contribution of W_mm into evt_part)
  event_kernel<<<B_, HEVT_, 0, stream>>>(marker, E, W_evt, b_evt, doy_idx,
                                         doy_mask, W_mm, evt_part);

  // 4) output heads
  {
    int total = B_ * N_ * PRED_;
    final_kernel<<<(total + 255) / 256, 256, 0, stream>>>(
        x, evt_part, W_mm, b_mm, W_ts, b_ts, (float*)d_out);
  }
}
